// GAT_83373905150065
// MI455X (gfx1250) — matrix operand order repaired and checked by reading, not compiled, not run on hardware
//
#include <hip/hip_runtime.h>
#include <hip/hip_bf16.h>

// GAT on MI455X (gfx1250). bf16 WMMA for GEMMs, f32 softmax/epilogues.
// B=1024, N=116 (pad 128), NFEAT=320, NHID=64, NHEADS=5, NCLASS=2.

#define GB 1024
#define GN 116
#define GNP 128
#define GF 320
#define GH 64
#define GHEADS 5
#define GC 2
#define NPAIR 6          // 5 heads + output-attention layer
#define LRELU 0.2f
#define NEGINF -9e15f

typedef __attribute__((ext_vector_type(16))) __bf16 bf16x16;
typedef __attribute__((ext_vector_type(8)))  __bf16 bf16x8;
typedef __attribute__((ext_vector_type(8)))  float  f32x8;

__device__ __forceinline__ bf16x16 cat8(bf16x8 lo, bf16x8 hi) {
  bf16x16 r;
#pragma unroll
  for (int i = 0; i < 8; ++i) { r[i] = lo[i]; r[8 + i] = hi[i]; }
  return r;
}

__device__ __forceinline__ f32x8 wmma_bf16(bf16x16 a, bf16x16 b, f32x8 c) {
  return __builtin_amdgcn_wmma_f32_16x16x32_bf16(false, a, false, b, (short)0, c,
                                                 false, false);
}

// B-fragment (K=32 x N=16) from row-major [col][k] bf16 storage.
// ISA: lanes 0-15 hold K={0..7,16..23} (col = lane), lanes 16-31 hold K={8..15,24..31}.
__device__ __forceinline__ bf16x16 load_b_rowmajor(const __bf16* base, int ld,
                                                   int n0, int k, int lane) {
  const __bf16* p = base + (size_t)(n0 + (lane & 15)) * ld + k + ((lane >> 4) << 3);
  bf16x8 lo = *(const bf16x8*)p;
  bf16x8 hi = *(const bf16x8*)(p + 16);
  return cat8(lo, hi);
}

// ---------------------------------------------------------------------------
// K0: wsrc[p][f] = sum_o W[p][f][o]*a[p][o], wdst with a[p][O+o]; slot 5 = out layer
__global__ void k_wvec(const float* __restrict__ Wh, const float* __restrict__ ah,
                       const float* __restrict__ Wo, const float* __restrict__ ao,
                       float* __restrict__ wsrc, float* __restrict__ wdst) {
  int i = blockIdx.x * blockDim.x + threadIdx.x;
  if (i >= NPAIR * GF) return;
  int p = i / GF, f = i % GF;
  float s = 0.f, d = 0.f;
  if (p < GHEADS) {
    const float* Wp = Wh + ((size_t)p * GF + f) * GH;
    const float* ap = ah + (size_t)p * 2 * GH;
    for (int o = 0; o < GH; ++o) { s += Wp[o] * ap[o]; d += Wp[o] * ap[GH + o]; }
  } else {
    const float* Wp = Wo + (size_t)f * GC;
    for (int c = 0; c < GC; ++c) { s += Wp[c] * ao[c]; d += Wp[c] * ao[GC + c]; }
  }
  wsrc[i] = s; wdst[i] = d;
}

// K0b: Wt[col][k] = bf16( W_heads[h][k][o] ), col = h*64+o
__global__ void k_wt(const float* __restrict__ Wh, __bf16* __restrict__ wt) {
  int i = blockIdx.x * blockDim.x + threadIdx.x;
  if (i >= GF * GF) return;
  int col = i / GF, k = i % GF;
  int h = col / GH, o = col % GH;
  wt[i] = (__bf16)Wh[((size_t)h * GF + k) * GH + o];
}

// ---------------------------------------------------------------------------
// Edge scores: one wave per (b,n) row; esrc/edst[row][p] = feat_row . wvec[p]
__global__ __launch_bounds__(256) void k_scores(const float* __restrict__ feat,
                                                const float* __restrict__ wsrc,
                                                const float* __restrict__ wdst,
                                                float* __restrict__ esrc,
                                                float* __restrict__ edst) {
  __shared__ float ws[NPAIR * GF];
  __shared__ float wd[NPAIR * GF];
  for (int i = threadIdx.x; i < NPAIR * GF; i += blockDim.x) {
    ws[i] = wsrc[i]; wd[i] = wdst[i];
  }
  __syncthreads();
  const int wave = threadIdx.x >> 5, lane = threadIdx.x & 31;
  size_t row = (size_t)blockIdx.x * 8 + wave;
  if (row >= (size_t)GB * GN) return;
  const float* fr = feat + row * GF;
  float as[NPAIR] = {}, ad[NPAIR] = {};
  for (int f = lane; f < GF; f += 32) {
    float v = fr[f];
#pragma unroll
    for (int p = 0; p < NPAIR; ++p) { as[p] += v * ws[p * GF + f]; ad[p] += v * wd[p * GF + f]; }
  }
#pragma unroll
  for (int off = 16; off > 0; off >>= 1) {
#pragma unroll
    for (int p = 0; p < NPAIR; ++p) {
      as[p] += __shfl_down(as[p], off);
      ad[p] += __shfl_down(ad[p], off);
    }
  }
  if (lane == 0) {
#pragma unroll
    for (int p = 0; p < NPAIR; ++p) { esrc[row * NPAIR + p] = as[p]; edst[row * NPAIR + p] = ad[p]; }
  }
}

// ---------------------------------------------------------------------------
// Fused GAT layer 1, one workgroup per (b, head), 8 waves.
__global__ __launch_bounds__(256) void k_gat1(const float* __restrict__ x,
                                              const int* __restrict__ adj,
                                              const __bf16* __restrict__ wt,
                                              const float* __restrict__ esrc,
                                              const float* __restrict__ edst,
                                              float* __restrict__ x1) {
  __shared__ __bf16 whT[GH][GNP + 8];    // Wh transposed: [feature][node], bf16
  __shared__ __bf16 attL[GNP][GNP + 8];  // unnormalized softmax probs, bf16
  __shared__ float edstL[GNP];
  __shared__ float rinv[GNP];

  const int b = blockIdx.x / GHEADS;
  const int h = blockIdx.x % GHEADS;
  const int tid = threadIdx.x;
  const int wave = tid >> 5;
  const int lane = tid & 31;
  const float* xb = x + (size_t)b * GN * GF;

  // ---- Stage 1: Wh = x[b] @ W_h via WMMA; store transposed bf16 into whT ----
  {
    f32x8 acc[4] = {};
    const int r = lane & 15;
    const int chunk = (lane >> 4) << 3;  // A layout: lanes<16 K{0..7,16..23}, else {8..15,24..31}
    int row = wave * 16 + r; if (row > GN - 1) row = GN - 1;   // clamp pad rows
    const float* arow = xb + (size_t)row * GF;
    for (int k = 0; k < GF; k += 32) {
      bf16x16 a;
      const float* p = arow + k + chunk;
#pragma unroll
      for (int i = 0; i < 8; ++i) { a[i] = (__bf16)p[i]; a[8 + i] = (__bf16)p[16 + i]; }
#pragma unroll
      for (int nt = 0; nt < 4; ++nt) {
        bf16x16 bf = load_b_rowmajor(wt, GF, h * GH + nt * 16, k, lane);
        acc[nt] = wmma_bf16(a, bf, acc[nt]);
      }
    }
#pragma unroll
    for (int nt = 0; nt < 4; ++nt) {
      int col = nt * 16 + (lane & 15);
#pragma unroll
      for (int rr = 0; rr < 8; ++rr) {
        int m = wave * 16 + rr + ((lane >> 4) << 3);   // C layout: rows r+8*(lane/16)
        whT[col][m] = (__bf16)acc[nt][rr];
      }
    }
  }
  if (tid < GNP)
    edstL[tid] = (tid < GN) ? edst[((size_t)b * GN + tid) * NPAIR + h] : 0.f;
  if (tid >= 128 && tid < 128 + (GNP - GN)) {          // zero pad rows 116..127
    int rr = GN + (tid - 128);
    for (int j = 0; j < GNP; ++j) attL[rr][j] = (__bf16)0.f;
  }
  __syncthreads();

  // ---- Phase A: masked leaky-relu + row softmax (unnormalized exp into LDS) ----
  if (tid < GN) {
    const int r = tid;
    const float es = esrc[((size_t)b * GN + r) * NPAIR + h];
    const int* arow = adj + ((size_t)b * GN + r) * GN;
    float mx = -3.4e38f;
    for (int j = 0; j < GN; ++j) {
      float e = es + edstL[j];
      e = (e > 0.f) ? e : LRELU * e;
      e = (arow[j] > 0) ? e : NEGINF;
      mx = fmaxf(mx, e);
    }
    float s = 0.f;
    for (int j = 0; j < GN; ++j) {
      float e = es + edstL[j];
      e = (e > 0.f) ? e : LRELU * e;
      e = (arow[j] > 0) ? e : NEGINF;
      float p = __expf(e - mx);
      s += p;
      attL[r][j] = (__bf16)p;
    }
    for (int j = GN; j < GNP; ++j) attL[r][j] = (__bf16)0.f;
    rinv[r] = 1.f / s;
  }
  __syncthreads();

  // ---- Phase B: h' = att @ Wh via WMMA from LDS; ELU epilogue; write x1 ----
  {
    f32x8 acc[4] = {};
    const int r = lane & 15;
    const int chunk = (lane >> 4) << 3;
    for (int k = 0; k < GNP; k += 32) {
      const __bf16* pa = &attL[wave * 16 + r][k + chunk];
      bf16x16 a = cat8(*(const bf16x8*)pa, *(const bf16x8*)(pa + 16));
#pragma unroll
      for (int nt = 0; nt < 4; ++nt) {
        int o = nt * 16 + (lane & 15);
        const __bf16* pb = &whT[o][k + ((lane >> 4) << 3)];
        bf16x16 bf = cat8(*(const bf16x8*)pb, *(const bf16x8*)(pb + 16));
        acc[nt] = wmma_bf16(a, bf, acc[nt]);
      }
    }
#pragma unroll
    for (int nt = 0; nt < 4; ++nt) {
      int col = h * GH + nt * 16 + (lane & 15);
#pragma unroll
      for (int rr = 0; rr < 8; ++rr) {
        int m = wave * 16 + rr + ((lane >> 4) << 3);
        if (m < GN) {
          float v = acc[nt][rr] * rinv[m];
          v = (v > 0.f) ? v : (__expf(v) - 1.f);     // ELU
          x1[((size_t)b * GN + m) * GF + col] = v;
        }
      }
    }
  }
}

// ---------------------------------------------------------------------------
// Pass-2 attentions -> d_out (atts region), LDS-staged for coalesced writes.
__global__ __launch_bounds__(128) void k_atts(const int* __restrict__ adj,
                                              const float* __restrict__ esrc,
                                              const float* __restrict__ edst,
                                              float* __restrict__ atts) {
  __shared__ float edstL[GN];
  __shared__ float rowL[GN][GN + 4];
  const int b = blockIdx.x / GHEADS;
  const int h = blockIdx.x % GHEADS;
  const int tid = threadIdx.x;
  if (tid < GN) edstL[tid] = edst[((size_t)b * GN + tid) * NPAIR + h];
  __syncthreads();
  if (tid < GN) {
    const int r = tid;
    const float es = esrc[((size_t)b * GN + r) * NPAIR + h];
    const int* arow = adj + ((size_t)b * GN + r) * GN;
    float mx = -3.4e38f;
    for (int j = 0; j < GN; ++j) {
      float e = es + edstL[j];
      e = (e > 0.f) ? e : LRELU * e;
      e = (arow[j] > 0) ? e : NEGINF;
      mx = fmaxf(mx, e);
    }
    float s = 0.f;
    for (int j = 0; j < GN; ++j) {
      float e = es + edstL[j];
      e = (e > 0.f) ? e : LRELU * e;
      e = (arow[j] > 0) ? e : NEGINF;
      float p = __expf(e - mx);
      s += p;
      rowL[r][j] = p;
    }
    float inv = 1.f / s;
    for (int j = 0; j < GN; ++j) rowL[r][j] *= inv;
  }
  __syncthreads();
  float* out = atts + ((size_t)h * GB + b) * GN * GN;
  for (int idx = tid; idx < GN * GN; idx += 128) {
    int r = idx / GN, j = idx - r * GN;
    out[idx] = rowL[r][j];
  }
}

// ---------------------------------------------------------------------------
// Output GAT layer (NCLASS=2) + final linear -> logits. One WG per batch b.
__global__ __launch_bounds__(128) void k_out(const float* __restrict__ x1,
                                             const int* __restrict__ adj,
                                             const float* __restrict__ esrc,
                                             const float* __restrict__ edst,
                                             const float* __restrict__ Wout,
                                             const float* __restrict__ Wlin,
                                             const float* __restrict__ blin,
                                             float* __restrict__ logits) {
  __shared__ float who[GNP][GC];
  __shared__ float edstL[GNP];
  __shared__ float x2s[GNP][GC];
  const int b = blockIdx.x;
  const int tid = threadIdx.x;
  if (tid < GNP) {
    int r = (tid < GN) ? tid : GN - 1;
    const float* xr = x1 + ((size_t)b * GN + r) * GF;
    float c0 = 0.f, c1 = 0.f;
    for (int f = 0; f < GF; ++f) { float v = xr[f]; c0 += v * Wout[f * GC]; c1 += v * Wout[f * GC + 1]; }
    who[tid][0] = c0; who[tid][1] = c1;
    edstL[tid] = (tid < GN) ? edst[((size_t)b * GN + tid) * NPAIR + 5] : 0.f;
  }
  __syncthreads();
  if (tid < GN) {
    const int r = tid;
    const float es = esrc[((size_t)b * GN + r) * NPAIR + 5];
    const int* arow = adj + ((size_t)b * GN + r) * GN;
    float mx = -3.4e38f;
    for (int j = 0; j < GN; ++j) {
      float e = es + edstL[j];
      e = (e > 0.f) ? e : LRELU * e;
      e = (arow[j] > 0) ? e : NEGINF;
      mx = fmaxf(mx, e);
    }
    float s = 0.f, a0 = 0.f, a1 = 0.f;
    for (int j = 0; j < GN; ++j) {
      float e = es + edstL[j];
      e = (e > 0.f) ? e : LRELU * e;
      e = (arow[j] > 0) ? e : NEGINF;
      float p = __expf(e - mx);
      s += p; a0 += p * who[j][0]; a1 += p * who[j][1];
    }
    float v0 = a0 / s, v1 = a1 / s;
    v0 = (v0 > 0.f) ? v0 : (__expf(v0) - 1.f);
    v1 = (v1 > 0.f) ? v1 : (__expf(v1) - 1.f);
    x2s[r][0] = v0; x2s[r][1] = v1;
  }
  __syncthreads();
  if (tid < GC) {
    float acc = blin[tid];
    const float* wl = Wlin + (size_t)tid * (GN * GC);
    const float* xf = &x2s[0][0];
    for (int j = 0; j < GN * GC; ++j) acc += xf[j] * wl[j];
    logits[(size_t)b * GC + tid] = acc;
  }
}

// ---------------------------------------------------------------------------
extern "C" void kernel_launch(void* const* d_in, const int* in_sizes, int n_in,
                              void* d_out, int out_size, void* d_ws, size_t ws_size,
                              hipStream_t stream) {
  (void)in_sizes; (void)n_in; (void)out_size; (void)ws_size;
  const float* x      = (const float*)d_in[0];
  const int*   adj    = (const int*)d_in[1];
  const float* Wheads = (const float*)d_in[2];
  const float* aheads = (const float*)d_in[3];
  const float* Wout   = (const float*)d_in[4];
  const float* aout   = (const float*)d_in[5];
  const float* Wlin   = (const float*)d_in[6];
  const float* blin   = (const float*)d_in[7];

  float* logits = (float*)d_out;                      // [B, 2]
  float* atts   = (float*)d_out + (size_t)GB * GC;    // [H, B, N, N]

  char* ws = (char*)d_ws;
  size_t off = 0;
  auto take = [&](size_t bytes) -> char* {
    char* p = ws + off;
    off += (bytes + 255) & ~(size_t)255;
    return p;
  };
  __bf16* wt  = (__bf16*)take((size_t)GF * GF * 2);
  float* wsrc = (float*)take((size_t)NPAIR * GF * 4);
  float* wdst = (float*)take((size_t)NPAIR * GF * 4);
  float* esrc = (float*)take((size_t)GB * GN * NPAIR * 4);
  float* edst = (float*)take((size_t)GB * GN * NPAIR * 4);
  float* x1   = (float*)take((size_t)GB * GN * GF * 4);

  k_wvec<<<(NPAIR * GF + 255) / 256, 256, 0, stream>>>(Wheads, aheads, Wout, aout, wsrc, wdst);
  k_wt<<<(GF * GF + 255) / 256, 256, 0, stream>>>(Wheads, wt);
  k_scores<<<(GB * GN + 7) / 8, 256, 0, stream>>>(x, wsrc, wdst, esrc, edst);
  k_gat1<<<GB * GHEADS, 256, 0, stream>>>(x, adj, wt, esrc, edst, x1);
  k_scores<<<(GB * GN + 7) / 8, 256, 0, stream>>>(x1, wsrc, wdst, esrc, edst);
  k_atts<<<GB * GHEADS, 128, 0, stream>>>(adj, esrc, edst, atts);
  k_out<<<GB, 128, 0, stream>>>(x1, adj, esrc, edst, Wout, Wlin, blin, logits);
}
